// SelfAttention_73967926771958
// MI455X (gfx1250) — matrix operand order repaired and checked by reading, not compiled, not run on hardware
//
#include <hip/hip_runtime.h>

// ---------- types ----------
typedef __attribute__((ext_vector_type(16))) __bf16        bf16x16;
typedef __attribute__((ext_vector_type(8)))  float         f32x8;
typedef __attribute__((ext_vector_type(4)))  unsigned int  u32x4;
typedef __attribute__((ext_vector_type(8)))  int           i32x8;
typedef __attribute__((ext_vector_type(4)))  int           i32x4;

union TileHalf { bf16x16 v; u32x4 q[2]; };

#define NB  2
#define NS  2048
#define ND  1024
#define NH  16
#define DH  64
#define NM  (NB * NS)          // 4096 rows of hidden

// ---------- helpers ----------
static __device__ __forceinline__ unsigned short f2bf(float f) {
    unsigned int u = __float_as_uint(f);
    u += 0x7FFFu + ((u >> 16) & 1u);          // round-to-nearest-even
    return (unsigned short)(u >> 16);
}

static __device__ __forceinline__ f32x8 wmma_bf16(bf16x16 a, bf16x16 b, f32x8 c) {
    return __builtin_amdgcn_wmma_f32_16x16x32_bf16(
        false, a, false, b, (short)0, c, false, false);
}

// A-operand 16x32 bf16: lane L holds row L%16; VGPR0-3 = K[hi*8 .. hi*8+7],
// VGPR4-7 = K[16+hi*8 .. 16+hi*8+7]  (hi = L/16), per ISA 7.12.2.
static __device__ __forceinline__ bf16x16
load_tile_a(const unsigned short* p, int ld, int rb, int kb, int lane) {
    int r = lane & 15, hi = lane >> 4;
    const unsigned short* base = p + (size_t)(rb + r) * ld + kb;
    TileHalf t;
    t.q[0] = *reinterpret_cast<const u32x4*>(base + hi * 8);
    t.q[1] = *reinterpret_cast<const u32x4*>(base + 16 + hi * 8);
    return t.v;
}

// B-operand 32x16 bf16 loaded from B^T rows (row-major [N][K]):
// lane L holds col N=L%16; VGPR0-3 = K[hi*8 .. hi*8+7], VGPR4-7 = K[16+hi*8 .. 16+hi*8+7].
static __device__ __forceinline__ bf16x16
load_tile_b(const unsigned short* p, int ld, int nb, int kb, int lane) {
    int n = lane & 15, hi = lane >> 4;
    const unsigned short* base = p + (size_t)(nb + n) * ld + kb;
    TileHalf t;
    t.q[0] = *reinterpret_cast<const u32x4*>(base + hi * 8);
    t.q[1] = *reinterpret_cast<const u32x4*>(base + 16 + hi * 8);
    return t.v;
}

// ---------- Tensor Data Mover: 2D tile (2-byte elements) global -> LDS ----------
// Descriptor per CDNA5 ISA (6-arg toolchain builtin).
static __device__ __forceinline__ void
tdm_load_2d_bf16(const unsigned short* gptr, unsigned lds_off,
                 int tile_x, int tile_y, int tensor_x, int tensor_y,
                 long long stride_x /* elements */) {
    unsigned long long ga = (unsigned long long)(size_t)gptr;
    u32x4 g0;
    g0[0] = 1u;                                           // count=1
    g0[1] = lds_off;                                      // lds_addr (bytes)
    g0[2] = (unsigned)(ga & 0xFFFFFFFFull);               // global_addr[31:0]
    g0[3] = (unsigned)((ga >> 32) & 0x1FFFFFFull) | (2u << 30);  // [120:96]|type=2
    i32x8 g1;
    g1[0] = 0x00010000;                                   // data_size=1 -> 2 bytes
    g1[1] = (int)((tensor_x & 0xFFFF) << 16);             // tensor_dim0[15:0]
    g1[2] = (int)(((unsigned)tensor_x >> 16) & 0xFFFFu) |
            (int)((tensor_y & 0xFFFF) << 16);             // dim0[31:16]|dim1[15:0]
    g1[3] = (int)(((unsigned)tensor_y >> 16) & 0xFFFFu) |
            (int)((tile_x & 0xFFFF) << 16);               // dim1[31:16]|tile_dim0
    g1[4] = (int)(tile_y & 0xFFFF);                       // tile_dim1 (tile_dim2=0)
    g1[5] = (int)(stride_x & 0xFFFFFFFFll);               // dim0_stride[31:0]
    g1[6] = (int)((stride_x >> 32) & 0xFFFFll);           // dim0_stride[47:32]
    g1[7] = 0;                                            // dim1_stride hi
    i32x4 z4 = {0, 0, 0, 0};
    i32x8 z8 = {0, 0, 0, 0, 0, 0, 0, 0};
    __builtin_amdgcn_tensor_load_to_lds(g0, g1, z4, z4, z8, 0);
}

static __device__ __forceinline__ unsigned lds_offset(const void* p) {
    return (unsigned)(size_t)p;    // generic ptr low 32 bits = LDS byte address
}

// ---------- kernel 1: f32 -> bf16 convert ----------
__global__ void k_cvt_bf16(const float* __restrict__ in,
                           unsigned short* __restrict__ out, int n) {
    int i = blockIdx.x * blockDim.x + threadIdx.x;
    if (i < n) out[i] = f2bf(in[i]);
}

// ---------- kernel 2: W (f32 [K][N]) -> W^T (bf16 [N][K]) ----------
__global__ void k_transpose_w(const float* __restrict__ wq,
                              const float* __restrict__ wk,
                              const float* __restrict__ wv,
                              unsigned short* __restrict__ outbase) {
    const float* in = (blockIdx.z == 0) ? wq : (blockIdx.z == 1) ? wk : wv;
    unsigned short* out = outbase + (size_t)blockIdx.z * ND * ND;
    __shared__ float tile[32][33];
    int x = blockIdx.x * 32 + threadIdx.x;
    int y = blockIdx.y * 32 + threadIdx.y;
    #pragma unroll
    for (int j = 0; j < 32; j += 8)
        tile[threadIdx.y + j][threadIdx.x] = in[(size_t)(y + j) * ND + x];
    __syncthreads();
    x = blockIdx.y * 32 + threadIdx.x;
    y = blockIdx.x * 32 + threadIdx.y;
    #pragma unroll
    for (int j = 0; j < 32; j += 8)
        out[(size_t)(y + j) * ND + x] = f2bf(tile[threadIdx.x][threadIdx.y + j]);
}

// ---------- kernel 3: QKV projection GEMM ----------
// Block = 8 waves = 128(M) x 64(N) tile; all waves share one n0, so the W^T
// 64x32 k-slice is TDM-prefetched into LDS (double buffered) by wave 0.
__global__ void __launch_bounds__(256)
k_qkv_gemm(const unsigned short* __restrict__ hb,
           const unsigned short* __restrict__ wt,   // 3 x [1024][1024] bf16 W^T
           const float* __restrict__ bq, const float* __restrict__ bk,
           const float* __restrict__ bv,
           unsigned short* __restrict__ qo, unsigned short* __restrict__ ko,
           unsigned short* __restrict__ vto) {
    __shared__ __align__(16) unsigned short ldsB[2][64 * 32];   // 8 KB

    int wslot = threadIdx.x >> 5;
    int lane  = threadIdx.x & 31;
    int wave  = blockIdx.x * 8 + wslot;
    int mat   = wave / (256 * 16);          // 0=q 1=k 2=v
    int t     = wave % (256 * 16);
    int m0    = (t % 256) * 16;
    int n0    = (t / 256) * 64;
    const unsigned short* w = wt + (size_t)mat * ND * ND;
    const float* bias = (mat == 0) ? bq : (mat == 1) ? bk : bv;

    if (wslot == 0) {  // preload k-slice 0: rows n0..n0+63, cols 0..31 of W^T
        tdm_load_2d_bf16(w + (size_t)n0 * ND, lds_offset(&ldsB[0][0]),
                         32, 64, ND, 64, ND);
        __builtin_amdgcn_s_wait_tensorcnt(0);
    }
    __syncthreads();

    f32x8 acc[4] = {};
    for (int k0 = 0; k0 < ND; k0 += 32) {
        int buf = (k0 >> 5) & 1;
        if (wslot == 0 && k0 + 32 < ND)
            tdm_load_2d_bf16(w + (size_t)n0 * ND + (k0 + 32),
                             lds_offset(&ldsB[buf ^ 1][0]), 32, 64, ND, 64, ND);
        __builtin_prefetch(hb + (size_t)(m0 + (lane & 15)) * ND + k0 + 128, 0, 1);
        bf16x16 a = load_tile_a(hb, ND, m0, k0, lane);
        const unsigned short* bp = &ldsB[buf][0];
        // batch all operand loads, then run the 4 WMMAs back-to-back
        bf16x16 b0 = load_tile_b(bp, 32, 0,  0, lane);
        bf16x16 b1 = load_tile_b(bp, 32, 16, 0, lane);
        bf16x16 b2 = load_tile_b(bp, 32, 32, 0, lane);
        bf16x16 b3 = load_tile_b(bp, 32, 48, 0, lane);
        acc[0] = wmma_bf16(a, b0, acc[0]);
        acc[1] = wmma_bf16(a, b1, acc[1]);
        acc[2] = wmma_bf16(a, b2, acc[2]);
        acc[3] = wmma_bf16(a, b3, acc[3]);
        if (wslot == 0) __builtin_amdgcn_s_wait_tensorcnt(0);
        __syncthreads();
    }

    int col = lane & 15, hi = lane >> 4;
    #pragma unroll
    for (int j = 0; j < 4; ++j) {
        int n = n0 + j * 16 + col;
        float bn = bias[n];
        int h = n >> 6, d = n & 63;
        #pragma unroll
        for (int rr = 0; rr < 8; ++rr) {
            int mg = m0 + rr + 8 * hi;
            int b  = mg >> 11, s = mg & (NS - 1);
            size_t bh = (size_t)(b * NH + h);
            unsigned short val = f2bf(acc[j][rr] + bn);
            if (mat == 0)      qo[(bh * NS + s) * DH + d] = val;
            else if (mat == 1) ko[(bh * NS + s) * DH + d] = val;
            else               vto[(bh * DH + d) * NS + s] = val;
        }
    }
}

// ---------- kernel 4: flash attention ----------
// Block = 8 waves = 8 consecutive 16-query tiles of ONE (b,h): all waves share
// the K / V^T stream, so both are TDM-prefetched into LDS (double buffered).
__global__ void __launch_bounds__(256)
k_attention(const unsigned short* __restrict__ q,
            const unsigned short* __restrict__ k,
            const unsigned short* __restrict__ vt,
            float* __restrict__ out) {
    __shared__ __align__(16) unsigned short ldsK[2][32 * 64];   // 8 KB
    __shared__ __align__(16) unsigned short ldsV[2][64 * 32];   // 8 KB
    __shared__ __align__(16) unsigned short pshare[8][16 * 32]; // 8 KB

    int wslot = threadIdx.x >> 5;
    int lane  = threadIdx.x & 31;
    int wave  = blockIdx.x * 8 + wslot;
    int bh    = wave >> 7;            // 0..31 (uniform across block)
    int qt    = wave & 127;           // 0..127
    const unsigned short* qp = q  + (size_t)bh * NS * DH;
    const unsigned short* kp = k  + (size_t)bh * NS * DH;
    const unsigned short* vp = vt + (size_t)bh * DH * NS;

    int row = lane & 15, hi = lane >> 4;

    // Q A-operands for the two 32-wide k-steps of dh=64 (held in regs all loop)
    bf16x16 aq0 = load_tile_a(qp, DH, qt * 16, 0,  lane);
    bf16x16 aq1 = load_tile_a(qp, DH, qt * 16, 32, lane);

    f32x8 o[4] = {};
    float m[8], l[8];
    #pragma unroll
    for (int rr = 0; rr < 8; ++rr) { m[rr] = -3.0e38f; l[rr] = 0.0f; }

    unsigned short* pl = &pshare[wslot][0];

    // preload key-block 0 into buffer 0
    if (wslot == 0) {
        tdm_load_2d_bf16(kp, lds_offset(&ldsK[0][0]), DH, 32, DH, NS, DH);
        tdm_load_2d_bf16(vp, lds_offset(&ldsV[0][0]), 32, DH, NS, DH, NS);
        __builtin_amdgcn_s_wait_tensorcnt(0);
    }
    __syncthreads();

    for (int j = 0; j < NS; j += 32) {
        int buf = (j >> 5) & 1;
        if (wslot == 0 && j + 32 < NS) {   // prefetch next key block
            tdm_load_2d_bf16(kp + (size_t)(j + 32) * DH,
                             lds_offset(&ldsK[buf ^ 1][0]), DH, 32, DH, NS, DH);
            tdm_load_2d_bf16(vp + (j + 32),
                             lds_offset(&ldsV[buf ^ 1][0]), 32, DH, NS, DH, NS);
        }
        const unsigned short* kl = &ldsK[buf][0];   // [32 keys][64] row-major
        const unsigned short* vl = &ldsV[buf][0];   // [64 d   ][32] row-major

        // ---- scores: batch 4 K-tile loads, then 4 back-to-back WMMAs ----
        f32x8 s0 = {}, s1 = {};
        {
            bf16x16 b00 = load_tile_b(kl, DH, 0,  0,  lane);
            bf16x16 b10 = load_tile_b(kl, DH, 16, 0,  lane);
            bf16x16 b01 = load_tile_b(kl, DH, 0,  32, lane);
            bf16x16 b11 = load_tile_b(kl, DH, 16, 32, lane);
            s0 = wmma_bf16(aq0, b00, s0);
            s1 = wmma_bf16(aq0, b10, s1);
            s0 = wmma_bf16(aq1, b01, s0);
            s1 = wmma_bf16(aq1, b11, s1);
        }
        // ---- streaming softmax update (scale 1/sqrt(64) = 0.125) ----
        #pragma unroll
        for (int rr = 0; rr < 8; ++rr) {
            float a0 = s0[rr] * 0.125f;
            float a1 = s1[rr] * 0.125f;
            float v = fmaxf(a0, a1);
            #pragma unroll
            for (int off = 8; off; off >>= 1)
                v = fmaxf(v, __shfl_xor(v, off, 16));
            float mn = fmaxf(m[rr], v);
            float alpha = __expf(m[rr] - mn);
            float p0 = __expf(a0 - mn);
            float p1 = __expf(a1 - mn);
            float sum = p0 + p1;
            #pragma unroll
            for (int off = 8; off; off >>= 1)
                sum += __shfl_xor(sum, off, 16);
            m[rr] = mn;
            l[rr] = l[rr] * alpha + sum;
            s0[rr] = p0;
            s1[rr] = p1;
            #pragma unroll
            for (int t = 0; t < 4; ++t) o[t][rr] *= alpha;
        }
        // ---- stage P (C-layout) into LDS as bf16, reload in A-layout ----
        int col = lane & 15;
        #pragma unroll
        for (int rr = 0; rr < 8; ++rr) {
            int rm = rr + 8 * hi;
            pl[rm * 32 + col]      = f2bf(s0[rr]);
            pl[rm * 32 + 16 + col] = f2bf(s1[rr]);
        }
        asm volatile("s_wait_dscnt 0" ::: "memory");  // own-wave LDS RAW fence
        TileHalf ap;
        ap.q[0] = *reinterpret_cast<const u32x4*>(pl + row * 32 + hi * 8);
        ap.q[1] = *reinterpret_cast<const u32x4*>(pl + row * 32 + 16 + hi * 8);
        // ---- ctx += P @ V : batch 4 V-tile loads, then 4 WMMAs ----
        {
            bf16x16 bv0 = load_tile_b(vl, 32, 0,  0, lane);
            bf16x16 bv1 = load_tile_b(vl, 32, 16, 0, lane);
            bf16x16 bv2 = load_tile_b(vl, 32, 32, 0, lane);
            bf16x16 bv3 = load_tile_b(vl, 32, 48, 0, lane);
            o[0] = wmma_bf16(ap.v, bv0, o[0]);
            o[1] = wmma_bf16(ap.v, bv1, o[1]);
            o[2] = wmma_bf16(ap.v, bv2, o[2]);
            o[3] = wmma_bf16(ap.v, bv3, o[3]);
        }
        if (wslot == 0) __builtin_amdgcn_s_wait_tensorcnt(0);
        __syncthreads();
    }

    // ---- normalize and store [B, S, H*64] ----
    int b = bh >> 4, h = bh & 15, col = lane & 15;
    #pragma unroll
    for (int rr = 0; rr < 8; ++rr) {
        float inv = 1.0f / l[rr];
        int s = qt * 16 + rr + 8 * hi;
        size_t base = ((size_t)(b * NS + s)) * ND + h * DH;
        #pragma unroll
        for (int t = 0; t < 4; ++t)
            out[base + t * 16 + col] = o[t][rr] * inv;
    }
}

// ---------- launcher ----------
extern "C" void kernel_launch(void* const* d_in, const int* in_sizes, int n_in,
                              void* d_out, int out_size, void* d_ws, size_t ws_size,
                              hipStream_t stream) {
    const float* hidden = (const float*)d_in[0];
    const float* Wq = (const float*)d_in[1];
    const float* bq = (const float*)d_in[2];
    const float* Wk = (const float*)d_in[3];
    const float* bk = (const float*)d_in[4];
    const float* Wv = (const float*)d_in[5];
    const float* bv = (const float*)d_in[6];
    float* out = (float*)d_out;

    char* ws = (char*)d_ws;
    const size_t HID_ELEMS = (size_t)NM * ND;          // 4,194,304
    unsigned short* hb  = (unsigned short*)(ws);                           // 8 MB
    unsigned short* wt  = (unsigned short*)(ws + HID_ELEMS * 2);           // 6 MB (3x W^T)
    unsigned short* qb  = (unsigned short*)(ws + HID_ELEMS * 2 + (size_t)3 * ND * ND * 2);
    unsigned short* kb  = qb + HID_ELEMS;                                  // 8 MB each
    unsigned short* vtb = kb + HID_ELEMS;

    // 1) hidden f32 -> bf16
    k_cvt_bf16<<<(int)(HID_ELEMS / 256), 256, 0, stream>>>(hidden, hb, (int)HID_ELEMS);

    // 2) W -> W^T bf16 (all three)
    k_transpose_w<<<dim3(32, 32, 3), dim3(32, 8), 0, stream>>>(Wq, Wk, Wv, wt);

    // 3) QKV projection: 3 * 256 * 16 wave-tiles, 8 waves/block
    k_qkv_gemm<<<(3 * 256 * 16) / 8, 256, 0, stream>>>(hb, wt, bq, bk, bv, qb, kb, vtb);

    // 4) flash attention: 32 bh * 128 q-tiles = 4096 waves, 8 waves/block
    k_attention<<<4096 / 8, 256, 0, stream>>>(qb, kb, vtb, out);
}
